// Complex2LayerMAPGraphConvolution_83391085019490
// MI455X (gfx1250) — matrix operand order repaired and checked by reading, not compiled, not run on hardware
//
#include <hip/hip_runtime.h>
#include <hip/hip_bf16.h>

// MI455X / gfx1250, wave32. D = A(16x32 f16) x B(32x16 f16) + C(16x16 f32).
typedef __attribute__((ext_vector_type(16))) _Float16 v16h;
typedef __attribute__((ext_vector_type(8)))  float    v8f;
typedef __attribute__((ext_vector_type(4)))  float    v4f;   // native vec4 (AS-safe)
typedef __attribute__((address_space(3))) const v4f* lds_cv4;

// ---------------------------------------------------------------------------
// Phase 1: per-edge complex weights  wr = w*cos(q*se), wi = w*sin(q*se)
// ---------------------------------------------------------------------------
__global__ __launch_bounds__(256) void edge_weights_kernel(
    const float* __restrict__ wsym, const float* __restrict__ q,
    const float* __restrict__ ent,  const float* __restrict__ cc,
    float* __restrict__ wr, float* __restrict__ wi, int E) {
  int e = blockIdx.x * blockDim.x + threadIdx.x;
  if (e >= E) return;
  float phase = q[0] * (ent[e] + cc[e]);
  float s, c;
  __sincosf(phase, &s, &c);
  float w = wsym[e];
  wr[e] = w * c;
  wi[e] = w * s;
}

// ---------------------------------------------------------------------------
// Phase 2: complex sparse scatter. 16 threads per edge, float4 (b128) gathers
// of x[col] (L2-resident: 25.6MB/matrix << 192MB L2), f32 atomic adds into
// aggR/aggI (also L2-resident).
//   aggR[row] += wr*xr[col] - wi*xi[col]
//   aggI[row] += wi*xr[col] + wr*xi[col]
// ---------------------------------------------------------------------------
__global__ __launch_bounds__(256) void scatter_kernel(
    const float* __restrict__ wr, const float* __restrict__ wi,
    const float* __restrict__ xr, const float* __restrict__ xi,
    const int* __restrict__ row, const int* __restrict__ col,
    float* __restrict__ aggR, float* __restrict__ aggI, int E) {
  long long t = (long long)blockIdx.x * blockDim.x + threadIdx.x;
  int e = (int)(t >> 4);
  if (e >= E) return;
  int fg = ((int)t & 15) << 2;               // feature group: 4 floats
  int c = col[e], r = row[e];
  float a = wr[e], b = wi[e];
  const v4f xv = *(const v4f*)(xr + (size_t)c * 64 + fg);
  const v4f yv = *(const v4f*)(xi + (size_t)c * 64 + fg);
  float* pR = aggR + (size_t)r * 64 + fg;
  float* pI = aggI + (size_t)r * 64 + fg;
  atomicAdd(pR + 0, a * xv.x - b * yv.x);
  atomicAdd(pR + 1, a * xv.y - b * yv.y);
  atomicAdd(pR + 2, a * xv.z - b * yv.z);
  atomicAdd(pR + 3, a * xv.w - b * yv.w);
  atomicAdd(pI + 0, b * xv.x + a * yv.x);
  atomicAdd(pI + 1, b * xv.y + a * yv.y);
  atomicAdd(pI + 2, b * xv.z + a * yv.z);
  atomicAdd(pI + 3, b * xv.w + a * yv.w);
}

// ---------------------------------------------------------------------------
// Async global->LDS b128 copy (CDNA5 GLOBAL_LOAD_ASYNC_TO_LDS_B128, ASYNCcnt,
// bypasses VGPRs). ldsaddr is an LDS byte address.
// ---------------------------------------------------------------------------
__device__ __forceinline__ void async_copy_b128(unsigned ldsaddr, const float* gp) {
  asm volatile("global_load_async_to_lds_b128 %0, %1, off"
               :: "v"(ldsaddr), "v"(gp) : "memory");
}
__device__ __forceinline__ void async_wait0() {
  asm volatile("s_wait_asynccnt 0" ::: "memory");
}

// ---------------------------------------------------------------------------
// WMMA fragment builders (layouts per CDNA5 ISA 7.12.2, wave32)
// A 16x32 f16: lanes 0-15 M=lane, K pairs 0..7 then 16..23; lanes 16-31 +8.
// Reads go through explicit addrspace(3) vec4 pointers -> ds_load_b128.
// ---------------------------------------------------------------------------
__device__ __forceinline__ v16h load_a_lds(unsigned tileBase, unsigned ldBytes,
                                           int koff, int lane) {
  int m = lane & 15;
  int kbase = koff + ((lane >> 4) << 3);     // +8 for lanes 16..31
  unsigned addr = tileBase + (unsigned)m * ldBytes + (unsigned)kbase * 4u;
  lds_cv4 p = (lds_cv4)(size_t)addr;
  v4f q0 = p[0];                             // K = kbase + 0..3
  v4f q1 = p[1];                             // K = kbase + 4..7
  v4f q2 = p[4];                             // K = kbase + 16..19
  v4f q3 = p[5];                             // K = kbase + 20..23
  v16h a;
  a[0]  = (_Float16)q0.x; a[1]  = (_Float16)q0.y; a[2]  = (_Float16)q0.z; a[3]  = (_Float16)q0.w;
  a[4]  = (_Float16)q1.x; a[5]  = (_Float16)q1.y; a[6]  = (_Float16)q1.z; a[7]  = (_Float16)q1.w;
  a[8]  = (_Float16)q2.x; a[9]  = (_Float16)q2.y; a[10] = (_Float16)q2.z; a[11] = (_Float16)q2.w;
  a[12] = (_Float16)q3.x; a[13] = (_Float16)q3.y; a[14] = (_Float16)q3.z; a[15] = (_Float16)q3.w;
  return a;
}

// B 32x16 f16 from row-major W[h][k] (B[k][n] = W[hoff+n][koff+k]):
// lanes 0-15 K=koff+0..7,16..23; lanes 16-31 K=koff+8..15,24..31; N = lane&15.
__device__ __forceinline__ v16h load_b_row(const float* Wp, int ldw, int hoff, int koff, int lane) {
  int n = lane & 15;
  int kbase = koff + ((lane >> 4) << 3);
  const float* p = Wp + (size_t)(hoff + n) * ldw + kbase;
  v16h b;
#pragma unroll
  for (int ee = 0; ee < 16; ++ee) b[ee] = (_Float16)p[ee < 8 ? ee : ee + 8];
  return b;
}

// ---------------------------------------------------------------------------
// Phase 3: Y[16-row tile] = relu(X @ W^T + bscale*bias). One wave per 16x64
// output tile. A-tile staged f32 into LDS via GLOBAL_LOAD_ASYNC_TO_LDS_B128;
// 2 A-frags reused across 4 h-tiles; 8 v_wmma_f32_16x16x32_f16 per wave.
// ---------------------------------------------------------------------------
__global__ __launch_bounds__(256) void gemm64_relu_kernel(
    const float* __restrict__ X, const float* __restrict__ W,
    const float* __restrict__ bias, float bscale,
    float* __restrict__ Y, int nrows) {
  __shared__ float lds[8 * 16 * 64];         // 32 KB: 4KB wave-private tile
  int wave = threadIdx.x >> 5;
  int lane = threadIdx.x & 31;
  int tile = blockIdx.x * 8 + wave;
  int ntiles = (nrows + 15) >> 4;
  if (tile >= ntiles) return;
  int row0 = tile << 4;
  bool full = (row0 + 16) <= nrows;          // wave-uniform
  // LDS byte address of this wave's tile: low 32 bits of the flat shared
  // pointer are the LDS aperture offset (ISA 10.2). Keeps `lds` alive too.
  unsigned txoff = (unsigned)(size_t)(lds + wave * (16 * 64));

#pragma unroll
  for (int i = lane; i < 256; i += 32) {     // 16 rows x 16 float4, coalesced
    int rr = i >> 4;
    int gr = row0 + rr; if (gr >= nrows) gr = nrows - 1;
    int kc = (i & 15) << 2;
    async_copy_b128(txoff + (unsigned)(rr * 64 + kc) * 4u,
                    X + (size_t)gr * 64 + kc);
  }
  async_wait0();

  v16h a0 = load_a_lds(txoff, 256, 0, lane);
  v16h a1 = load_a_lds(txoff, 256, 32, lane);
  int n = lane & 15;
  int mbase = (lane >> 4) << 3;              // C/D: VGPR g -> M = mbase+g
#pragma unroll
  for (int ht = 0; ht < 4; ++ht) {
    int hoff = ht << 4;
    v16h b0 = load_b_row(W, 64, hoff, 0, lane);
    v16h b1 = load_b_row(W, 64, hoff, 32, lane);
    v8f acc = {};
    acc = __builtin_amdgcn_wmma_f32_16x16x32_f16(false, a0, false, b0, (short)0, acc, false, false);
    acc = __builtin_amdgcn_wmma_f32_16x16x32_f16(false, a1, false, b1, (short)0, acc, false, false);
    float bb = bscale * bias[hoff + n];
    float* yp = Y + (size_t)(row0 + mbase) * 64 + hoff + n;
    if (full) {                              // uniform branch, no exec churn
#pragma unroll
      for (int g = 0; g < 8; ++g) {
        float v = acc[g] + bb;
        yp[(size_t)g * 64] = v > 0.f ? v : 0.f;   // stores w/ immediate offsets
      }
    } else {
#pragma unroll
      for (int g = 0; g < 8; ++g) {
        if (row0 + mbase + g < nrows) {
          float v = acc[g] + bb;
          yp[(size_t)g * 64] = v > 0.f ? v : 0.f;
        }
      }
    }
  }
}

// ---------------------------------------------------------------------------
// Phase 4: out = concat(Xr, Xi) @ W3^T + b3.  K=128 in 4 WMMA steps, one wave
// per 16x16 output tile; the concat happens in LDS at async-staging time.
// ---------------------------------------------------------------------------
__global__ __launch_bounds__(256) void gemm_out_kernel(
    const float* __restrict__ Xr, const float* __restrict__ Xi,
    const float* __restrict__ W3, const float* __restrict__ b3,
    float* __restrict__ out, int nrows) {
  __shared__ float lds[8 * 16 * 128];        // 64 KB: 8KB wave-private tile
  int wave = threadIdx.x >> 5;
  int lane = threadIdx.x & 31;
  int tile = blockIdx.x * 8 + wave;
  int ntiles = (nrows + 15) >> 4;
  if (tile >= ntiles) return;
  int row0 = tile << 4;
  bool full = (row0 + 16) <= nrows;
  unsigned txoff = (unsigned)(size_t)(lds + wave * (16 * 128));

#pragma unroll
  for (int i = lane; i < 256; i += 32) {
    int rr = i >> 4;
    int gr = row0 + rr; if (gr >= nrows) gr = nrows - 1;
    int kc = (i & 15) << 2;
    async_copy_b128(txoff + (unsigned)(rr * 128 + kc) * 4u,
                    Xr + (size_t)gr * 64 + kc);
    async_copy_b128(txoff + (unsigned)(rr * 128 + 64 + kc) * 4u,
                    Xi + (size_t)gr * 64 + kc);
  }
  async_wait0();

  v8f acc = {};
#pragma unroll
  for (int ks = 0; ks < 4; ++ks) {
    int kg = ks << 5;
    v16h a = load_a_lds(txoff, 512, kg, lane);
    v16h b = load_b_row(W3, 128, 0, kg, lane);   // W3 is [16][128]
    acc = __builtin_amdgcn_wmma_f32_16x16x32_f16(false, a, false, b, (short)0, acc, false, false);
  }
  int n = lane & 15;
  int mbase = (lane >> 4) << 3;
  float bb = b3[n];
  float* op = out + (size_t)(row0 + mbase) * 16 + n;
  if (full) {
#pragma unroll
    for (int g = 0; g < 8; ++g) op[(size_t)g * 16] = acc[g] + bb;
  } else {
#pragma unroll
    for (int g = 0; g < 8; ++g)
      if (row0 + mbase + g < nrows) op[(size_t)g * 16] = acc[g] + bb;
  }
}

// ---------------------------------------------------------------------------
extern "C" void kernel_launch(void* const* d_in, const int* in_sizes, int n_in,
                              void* d_out, int out_size, void* d_ws, size_t ws_size,
                              hipStream_t stream) {
  const float* xr   = (const float*)d_in[0];   // real_feature [N,64]
  const float* xi   = (const float*)d_in[1];   // imag_feature [N,64]
  const float* wsym = (const float*)d_in[2];   // [E]
  const float* q    = (const float*)d_in[3];   // scalar
  const float* ent  = (const float*)d_in[4];   // [E]
  const float* cc   = (const float*)d_in[5];   // [E]
  const float* W1   = (const float*)d_in[6];   // [64,64]
  const float* b1   = (const float*)d_in[7];   // [64]
  const float* W2   = (const float*)d_in[8];   // [64,64]
  const float* b2   = (const float*)d_in[9];   // [64]
  const float* W3   = (const float*)d_in[10];  // [16,128]
  const float* b3   = (const float*)d_in[11];  // [16]
  const int*   row  = (const int*)d_in[12];    // [E]
  const int*   col  = (const int*)d_in[13];    // [E]
  float* out = (float*)d_out;

  const int N = in_sizes[0] / 64;
  const int E = in_sizes[2];
  const size_t NH = (size_t)N * 64;

  // scratch layout (floats): wr | wi | aggR | aggI | actR | actI  (~110 MB)
  float* ws   = (float*)d_ws;
  float* wr   = ws;
  float* wi   = wr + E;
  float* aggR = wi + E;
  float* aggI = aggR + NH;
  float* actR = aggI + NH;
  float* actI = actR + NH;

  const int ntiles = (N + 15) / 16;
  const int gblocks = (ntiles + 7) / 8;
  const int sblocks = (int)(((size_t)E * 16 + 255) / 256);

  // Phase 1: edge weights
  edge_weights_kernel<<<(E + 255) / 256, 256, 0, stream>>>(wsym, q, ent, cc, wr, wi, E);

  // ---- Layer 1 ----
  (void)hipMemsetAsync(aggR, 0, 2 * NH * sizeof(float), stream);  // aggR+aggI contiguous
  scatter_kernel<<<sblocks, 256, 0, stream>>>(wr, wi, xr, xi, row, col, aggR, aggI, E);
  gemm64_relu_kernel<<<gblocks, 256, 0, stream>>>(aggR, W1, b1, 0.0f, actR, N);  // out_r: bias cancels
  gemm64_relu_kernel<<<gblocks, 256, 0, stream>>>(aggI, W1, b1, 2.0f, actI, N);  // out_i: bias doubles

  // ---- Layer 2 ----
  (void)hipMemsetAsync(aggR, 0, 2 * NH * sizeof(float), stream);
  scatter_kernel<<<sblocks, 256, 0, stream>>>(wr, wi, actR, actI, row, col, aggR, aggI, E);
  gemm64_relu_kernel<<<gblocks, 256, 0, stream>>>(aggR, W2, b2, 0.0f, actR, N);  // overwrite dead L1 acts
  gemm64_relu_kernel<<<gblocks, 256, 0, stream>>>(aggI, W2, b2, 2.0f, actI, N);

  // ---- Output head ----
  gemm_out_kernel<<<gblocks, 256, 0, stream>>>(actR, actI, W3, b3, out, N);
}
